// ParaGCNXBN00_89807766159501
// MI455X (gfx1250) — matrix operand order repaired and checked by reading, not compiled, not run on hardware
//
#include <hip/hip_runtime.h>
#include <hip/hip_bf16.h>

typedef __attribute__((ext_vector_type(16))) __bf16 v16bf;
typedef __attribute__((ext_vector_type(8)))  __bf16 v8bf;
typedef __attribute__((ext_vector_type(8)))  float  v8f;

#define LEAK 0.2f
#define EPS  1e-5f

static inline int cdiv(int a, int b) { return (a + b - 1) / b; }

// ---------------------------------------------------------------------------
// fp32 -> bf16 convert (activations)
__global__ void k_f32_to_bf16(const float* __restrict__ in, __bf16* __restrict__ out, long n) {
    long i = (long)blockIdx.x * blockDim.x + threadIdx.x;
    if (i < n) out[i] = (__bf16)in[i];
}

// Wt[m][k] = (m < M) ? W[k][m] : 0   (bf16, M padded to Mpad)
__global__ void k_transpose_w(const float* __restrict__ W, __bf16* __restrict__ Wt,
                              int K, int M, int Mpad) {
    long i = (long)blockIdx.x * blockDim.x + threadIdx.x;
    if (i >= (long)Mpad * K) return;
    int m = (int)(i / K), k = (int)(i % K);
    Wt[i] = (m < M) ? (__bf16)W[(long)k * M + m] : (__bf16)0.0f;
}

__global__ void k_zero(float* __restrict__ p, long n) {
    long i = (long)blockIdx.x * blockDim.x + threadIdx.x;
    if (i < n) p[i] = 0.0f;
}

// ---------------------------------------------------------------------------
// C[N,Mout] = A[N,K](bf16) @ Bt[Mpad,K]^T(bf16), fp32 accum via v_wmma_f32_16x16x32_bf16.
// 8 waves/block, wave -> one 16x16 tile; block covers 32 rows x 64 cols.
__global__ __launch_bounds__(256)
void k_gemm_bf16_wmma(const __bf16* __restrict__ A, const __bf16* __restrict__ Bt,
                      float* __restrict__ C, int N, int K, int Mout) {
    const int wave = threadIdx.x >> 5;
    const int lane = threadIdx.x & 31;
    const int row_tile = blockIdx.y * 2 + (wave >> 2);
    const int col_tile = blockIdx.x * 4 + (wave & 3);
    if (row_tile * 16 >= N) return;                      // wave-uniform

    const int m = row_tile * 16 + (lane & 15);
    const int n = col_tile * 16 + (lane & 15);
    const int khalfA = (lane >> 4) * 16;                 // A: K contiguous 16
    const int khalfB = (lane >> 4) * 16;                 // B: K contiguous 16
    const __bf16* arow = A + (size_t)m * K;
    const __bf16* brow = Bt + (size_t)n * K;

    v8f acc = {};
    for (int k0 = 0; k0 < K; k0 += 32) {
        v8bf alo = *(const v8bf*)(arow + k0 + khalfA);
        v8bf ahi = *(const v8bf*)(arow + k0 + 8 + khalfA);
        v16bf av;
        #pragma unroll
        for (int i = 0; i < 8; ++i) { av[i] = alo[i]; av[i + 8] = ahi[i]; }
        v16bf bv = *(const v16bf*)(brow + k0 + khalfB);
        acc = __builtin_amdgcn_wmma_f32_16x16x32_bf16(false, av, false, bv,
                                                      (short)0, acc, false, false);
    }
    // D layout: VGPR r -> row m = r + 8*(lane/16), col n = lane%16
    const int mbase = row_tile * 16 + (lane >> 4) * 8;
    if (n < Mout) {
        #pragma unroll
        for (int r = 0; r < 8; ++r)
            C[(size_t)(mbase + r) * Mout + n] = acc[r];
    }
}

// ---------------------------------------------------------------------------
// s_src[n][h] = sum_f h[n][h*32+f] * a[h][f];  s_dst with a[h][32+f]   (H=8, Fo=32, D=256)
__global__ void k_scores_h8(const float* __restrict__ Hl, const float* __restrict__ a,
                            float* __restrict__ Sa, float* __restrict__ Sb, int N) {
    long gid = (long)blockIdx.x * blockDim.x + threadIdx.x;
    if (gid >= (long)N * 8) return;
    int n = (int)(gid >> 3), h = (int)(gid & 7);
    const float* hp = Hl + (size_t)n * 256 + h * 32;
    const float* as = a + h * 64;
    const float* ad = as + 32;
    float sa = 0.f, sb = 0.f;
    #pragma unroll
    for (int f = 0; f < 32; ++f) { sa += hp[f] * as[f]; sb += hp[f] * ad[f]; }
    Sa[gid] = sa; Sb[gid] = sb;
}

// one wave per edge: 8-head softmax + alpha-weighted gather/scatter of 256 feats
__global__ __launch_bounds__(256)
void k_edge_h8(const int* __restrict__ src, const int* __restrict__ dst, int E,
               const float* __restrict__ Sa, const float* __restrict__ Sb,
               const float* __restrict__ Hl, float* __restrict__ Agg) {
    const int wave = threadIdx.x >> 5, lane = threadIdx.x & 31;
    long e = (long)blockIdx.x * 8 + wave;
    if (e >= E) return;
    int s = src[e], d = dst[e];
    float ev[8], mx = -1e30f;
    #pragma unroll
    for (int h = 0; h < 8; ++h) {
        float t = Sa[(size_t)s * 8 + h] + Sb[(size_t)d * 8 + h];
        t = t > 0.f ? t : LEAK * t;
        ev[h] = t; mx = fmaxf(mx, t);
    }
    float sum = 0.f;
    #pragma unroll
    for (int h = 0; h < 8; ++h) { ev[h] = __expf(ev[h] - mx); sum += ev[h]; }
    float inv = 1.0f / sum;
    const float* hs = Hl + (size_t)s * 256;
    float* od = Agg + (size_t)d * 256;
    #pragma unroll
    for (int i = 0; i < 8; ++i) {            // chunk i == head i (Fo==32)
        float al = ev[i] * inv;
        atomicAdd(od + i * 32 + lane, al * hs[i * 32 + lane]);
    }
}

// self-loops with multiplicity (src == dst == n)
__global__ __launch_bounds__(256)
void k_self_h8(const float* __restrict__ Sa, const float* __restrict__ Sb,
               const float* __restrict__ Hl, float* __restrict__ Agg, int N, float mult) {
    const int wave = threadIdx.x >> 5, lane = threadIdx.x & 31;
    long n = (long)blockIdx.x * 8 + wave;
    if (n >= N) return;
    float ev[8], mx = -1e30f;
    #pragma unroll
    for (int h = 0; h < 8; ++h) {
        float t = Sa[(size_t)n * 8 + h] + Sb[(size_t)n * 8 + h];
        t = t > 0.f ? t : LEAK * t;
        ev[h] = t; mx = fmaxf(mx, t);
    }
    float sum = 0.f;
    #pragma unroll
    for (int h = 0; h < 8; ++h) { ev[h] = __expf(ev[h] - mx); sum += ev[h]; }
    float inv = mult / sum;
    const float* hs = Hl + (size_t)n * 256;
    float* od = Agg + (size_t)n * 256;
    #pragma unroll
    for (int i = 0; i < 8; ++i)
        atomicAdd(od + i * 32 + lane, ev[i] * inv * hs[i * 32 + lane]);
}

// layer 3: H=1 -> softmax over a single head == 1; plain scatter-add of h[src]
__global__ __launch_bounds__(256)
void k_edge_h1(const int* __restrict__ src, const int* __restrict__ dst, int E,
               const float* __restrict__ Hl, float* __restrict__ Agg) {
    const int wave = threadIdx.x >> 5, lane = threadIdx.x & 31;
    long e = (long)blockIdx.x * 8 + wave;
    if (e >= E) return;
    int s = src[e], d = dst[e];
    for (int f = lane; f < 40; f += 32)
        atomicAdd(Agg + (size_t)d * 40 + f, Hl[(size_t)s * 40 + f]);
}

// ---------------------------------------------------------------------------
// in-place LayerNorm (+ optional ELU), D=256, one wave per node
__global__ __launch_bounds__(256)
void k_ln_elu_256(float* __restrict__ X, const float* __restrict__ g,
                  const float* __restrict__ b, int N, int do_elu) {
    const int wave = threadIdx.x >> 5, lane = threadIdx.x & 31;
    long n = (long)blockIdx.x * 8 + wave;
    if (n >= N) return;
    float* row = X + (size_t)n * 256;
    float v[8], s = 0.f;
    #pragma unroll
    for (int i = 0; i < 8; ++i) { v[i] = row[i * 32 + lane]; s += v[i]; }
    #pragma unroll
    for (int o = 16; o > 0; o >>= 1) s += __shfl_xor(s, o, 32);
    float mu = s * (1.0f / 256.0f);
    float s2 = 0.f;
    #pragma unroll
    for (int i = 0; i < 8; ++i) { float t = v[i] - mu; s2 += t * t; }
    #pragma unroll
    for (int o = 16; o > 0; o >>= 1) s2 += __shfl_xor(s2, o, 32);
    float r = rsqrtf(s2 * (1.0f / 256.0f) + EPS);
    #pragma unroll
    for (int i = 0; i < 8; ++i) {
        int f = i * 32 + lane;
        float y = (v[i] - mu) * r * g[f] + b[f];
        if (do_elu) y = y > 0.f ? y : (__expf(y) - 1.0f);
        row[f] = y;
    }
}

// final: LN over D=40 of (Agg + 3*H3) -> out, one wave per node
__global__ __launch_bounds__(256)
void k_ln40_out(const float* __restrict__ Agg, const float* __restrict__ H3,
                const float* __restrict__ g, const float* __restrict__ b,
                float* __restrict__ Out, int N) {
    const int wave = threadIdx.x >> 5, lane = threadIdx.x & 31;
    long n = (long)blockIdx.x * 8 + wave;
    if (n >= N) return;
    const float* ar = Agg + (size_t)n * 40;
    const float* hr = H3 + (size_t)n * 40;
    int f0 = lane, f1 = lane + 32;
    float x0 = (f0 < 40) ? ar[f0] + 3.0f * hr[f0] : 0.f;
    float x1 = (f1 < 40) ? ar[f1] + 3.0f * hr[f1] : 0.f;
    float s = x0 + x1;
    #pragma unroll
    for (int o = 16; o > 0; o >>= 1) s += __shfl_xor(s, o, 32);
    float mu = s * (1.0f / 40.0f);
    float d0 = (f0 < 40) ? x0 - mu : 0.f;
    float d1 = (f1 < 40) ? x1 - mu : 0.f;
    float s2 = d0 * d0 + d1 * d1;
    #pragma unroll
    for (int o = 16; o > 0; o >>= 1) s2 += __shfl_xor(s2, o, 32);
    float r = rsqrtf(s2 * (1.0f / 40.0f) + EPS);
    if (f0 < 40) Out[(size_t)n * 40 + f0] = d0 * r * g[f0] + b[f0];
    if (f1 < 40) Out[(size_t)n * 40 + f1] = d1 * r * g[f1] + b[f1];
}

// ---------------------------------------------------------------------------
extern "C" void kernel_launch(void* const* d_in, const int* in_sizes, int n_in,
                              void* d_out, int out_size, void* d_ws, size_t ws_size,
                              hipStream_t stream) {
    const int N = 50000, NFEAT = 128, D1 = 256, NCLASS = 40;
    const int E = in_sizes[1] / 2;

    const float* x  = (const float*)d_in[0];
    const int* src  = (const int*)d_in[1];
    const int* dst  = src + E;
    const float* W1 = (const float*)d_in[2];  const float* a1 = (const float*)d_in[3];
    const float* W2 = (const float*)d_in[4];  const float* a2 = (const float*)d_in[5];
    const float* W3 = (const float*)d_in[6];
    const float* g1 = (const float*)d_in[8];  const float* b1 = (const float*)d_in[9];
    const float* g2 = (const float*)d_in[10]; const float* b2 = (const float*)d_in[11];
    const float* g3 = (const float*)d_in[12]; const float* b3 = (const float*)d_in[13];
    float* out = (float*)d_out;

    // workspace carve-up (256B aligned)
    char* wsp = (char*)d_ws; size_t off = 0;
    auto carve = [&](size_t bytes) { void* p = wsp + off; off += (bytes + 255) & ~(size_t)255; return p; };
    __bf16* Xbf  = (__bf16*)carve((size_t)N * 256 * sizeof(__bf16));
    __bf16* Wt   = (__bf16*)carve((size_t)256 * 256 * sizeof(__bf16));
    float*  Hlin = (float*) carve((size_t)N * 256 * sizeof(float));
    float*  Sa   = (float*) carve((size_t)N * 8 * sizeof(float));
    float*  Sb   = (float*) carve((size_t)N * 8 * sizeof(float));
    float*  Agg  = (float*) carve((size_t)N * 256 * sizeof(float));

    const int TB = 256;
    const int row_blocks = cdiv(N / 16, 2);   // 16-row tiles, 2 per block

    // ---------------- layer 1 ----------------
    k_f32_to_bf16<<<cdiv(N * NFEAT, TB), TB, 0, stream>>>(x, Xbf, (long)N * NFEAT);
    k_transpose_w<<<cdiv(D1 * NFEAT, TB), TB, 0, stream>>>(W1, Wt, NFEAT, D1, D1);
    k_gemm_bf16_wmma<<<dim3(D1 / 64, row_blocks), TB, 0, stream>>>(Xbf, Wt, Hlin, N, NFEAT, D1);
    k_scores_h8<<<cdiv(N * 8, TB), TB, 0, stream>>>(Hlin, a1, Sa, Sb, N);
    k_zero<<<cdiv(N * 256, TB), TB, 0, stream>>>(Agg, (long)N * 256);
    k_edge_h8<<<cdiv(E, 8), TB, 0, stream>>>(src, dst, E, Sa, Sb, Hlin, Agg);
    k_self_h8<<<cdiv(N, 8), TB, 0, stream>>>(Sa, Sb, Hlin, Agg, N, 1.0f);
    k_ln_elu_256<<<cdiv(N, 8), TB, 0, stream>>>(Agg, g1, b1, N, 1);

    // ---------------- layer 2 ----------------
    k_f32_to_bf16<<<cdiv(N * D1, TB), TB, 0, stream>>>(Agg, Xbf, (long)N * D1);
    k_transpose_w<<<cdiv(D1 * D1, TB), TB, 0, stream>>>(W2, Wt, D1, D1, D1);
    k_gemm_bf16_wmma<<<dim3(D1 / 64, row_blocks), TB, 0, stream>>>(Xbf, Wt, Hlin, N, D1, D1);
    k_scores_h8<<<cdiv(N * 8, TB), TB, 0, stream>>>(Hlin, a2, Sa, Sb, N);
    k_zero<<<cdiv(N * 256, TB), TB, 0, stream>>>(Agg, (long)N * 256);
    k_edge_h8<<<cdiv(E, 8), TB, 0, stream>>>(src, dst, E, Sa, Sb, Hlin, Agg);
    k_self_h8<<<cdiv(N, 8), TB, 0, stream>>>(Sa, Sb, Hlin, Agg, N, 2.0f);
    k_ln_elu_256<<<cdiv(N, 8), TB, 0, stream>>>(Agg, g2, b2, N, 1);

    // ---------------- layer 3 (H=1, softmax over 1 head == identity) --------
    k_f32_to_bf16<<<cdiv(N * D1, TB), TB, 0, stream>>>(Agg, Xbf, (long)N * D1);
    k_transpose_w<<<cdiv(64 * D1, TB), TB, 0, stream>>>(W3, Wt, D1, NCLASS, 64);
    k_gemm_bf16_wmma<<<dim3(1, row_blocks), TB, 0, stream>>>(Xbf, Wt, Hlin, N, D1, NCLASS);
    k_zero<<<cdiv(N * NCLASS, TB), TB, 0, stream>>>(Agg, (long)N * NCLASS);
    k_edge_h1<<<cdiv(E, 8), TB, 0, stream>>>(src, dst, E, Hlin, Agg);
    k_ln40_out<<<cdiv(N, 8), TB, 0, stream>>>(Agg, Hlin, g3, b3, out, N);
}
